// MultilayerGRU_38379827757724
// MI455X (gfx1250) — matrix operand order repaired and checked by reading, not compiled, not run on hardware
//
#include <hip/hip_runtime.h>
#include <hip/hip_bf16.h>

typedef __attribute__((ext_vector_type(16))) __bf16 v16bf;
typedef __attribute__((ext_vector_type(8)))  __bf16 v8bf;
typedef __attribute__((ext_vector_type(8)))  float  v8f;

#define B_ 64
#define S_ 1024
#define I_ 256
#define H_ 1024
#define O_ 256

static __device__ __forceinline__ v8bf ld8(const __bf16* p) {
  return *(const v8bf*)p;
}
static __device__ __forceinline__ v16bf cat8(v8bf lo, v8bf hi) {
  return __builtin_shufflevector(lo, hi, 0,1,2,3,4,5,6,7,8,9,10,11,12,13,14,15);
}
// Fast sigmoid/tanh: v_exp_f32 + raw v_rcp_f32 (no IEEE div fixup sequence).
static __device__ __forceinline__ float fast_sigmoid(float x) {
  return __builtin_amdgcn_rcpf(1.0f + __expf(-x));
}
static __device__ __forceinline__ float fast_tanh(float x) {
  return 2.0f * fast_sigmoid(2.0f * x) - 1.0f;
}

// ---------------------------------------------------------------------------
// Phase 1: acc = [h|x] @ [Whz;Wxz | Whr;Wxr]  (N = 2H).  z = sigmoid(acc+bhz),
// r = sigmoid(acc+bhr); store z (f32) and r*h (bf16) for phase 2.
// Grid (32,2), block 128 (4 waves). Wave -> one 32x16 output strip.
// K-loop split at the h/x boundary: no per-iteration operand selects.
// ---------------------------------------------------------------------------
template <int IND>
__global__ __launch_bounds__(128)
void gru_phase1_kernel(const __bf16* __restrict__ hb,
                       const __bf16* __restrict__ xb,
                       const __bf16* __restrict__ WzrT,   // [2H][H+IND] n-major
                       const float*  __restrict__ bhz,
                       const float*  __restrict__ bhr,
                       const float*  __restrict__ hf,
                       float*        __restrict__ zbuf,
                       __bf16*       __restrict__ rhbuf)
{
  constexpr int Ktot = H_ + IND;
  const int lane  = threadIdx.x & 31;
  const int wave  = threadIdx.x >> 5;
  const int ntile = blockIdx.x * 4 + wave;        // [0,128)
  const int nlane = ntile * 16 + (lane & 15);     // column in [0,2H)
  const int mbase = blockIdx.y * 32;
  const int hB    = 8  * (lane >> 4);             // B-frag K sub-offset
  const int hA    = 8  * (lane >> 4);             // A-frag K sub-offset
  const int m0    = mbase + (lane & 15);

  const __bf16* brow = WzrT + (size_t)nlane * Ktot + hB;
  const __bf16* ah0  = hb + (size_t)m0 * H_ + hA;
  const __bf16* ah1  = ah0 + 16 * H_;
  const __bf16* ax0  = xb + (size_t)m0 * IND + hA;
  const __bf16* ax1  = ax0 + 16 * IND;

  v8f acc0 = {};
  v8f acc1 = {};

  for (int kb = 0; kb < H_; kb += 32) {           // h-part of K
    const __bf16* bp = brow + kb;
    __builtin_prefetch(bp + 256, 0, 1);           // global_prefetch_b8
    v16bf bfrag = cat8(ld8(bp), ld8(bp + 16));
    v16bf a0 = cat8(ld8(ah0 + kb), ld8(ah0 + kb + 16));
    acc0 = __builtin_amdgcn_wmma_f32_16x16x32_bf16(
        false, a0, false, bfrag, (short)0, acc0, false, false);
    v16bf a1 = cat8(ld8(ah1 + kb), ld8(ah1 + kb + 16));
    acc1 = __builtin_amdgcn_wmma_f32_16x16x32_bf16(
        false, a1, false, bfrag, (short)0, acc1, false, false);
  }
  const __bf16* browx = brow + H_;
  for (int kb = 0; kb < IND; kb += 32) {          // x-part of K
    const __bf16* bp = browx + kb;
    v16bf bfrag = cat8(ld8(bp), ld8(bp + 16));
    v16bf a0 = cat8(ld8(ax0 + kb), ld8(ax0 + kb + 16));
    acc0 = __builtin_amdgcn_wmma_f32_16x16x32_bf16(
        false, a0, false, bfrag, (short)0, acc0, false, false);
    v16bf a1 = cat8(ld8(ax1 + kb), ld8(ax1 + kb + 16));
    acc1 = __builtin_amdgcn_wmma_f32_16x16x32_bf16(
        false, a1, false, bfrag, (short)0, acc1, false, false);
  }

  const bool is_z = (ntile < 64);
  const int  ncol = is_z ? nlane : (nlane - H_);
  const float bias = is_z ? bhz[ncol] : bhr[ncol];
#pragma unroll
  for (int e = 0; e < 8; ++e) {
    const int r0 = mbase + (lane >> 4) * 8 + e;
    const int r1 = r0 + 16;
    if (is_z) {
      zbuf[r0 * H_ + ncol] = fast_sigmoid(acc0[e] + bias);
      zbuf[r1 * H_ + ncol] = fast_sigmoid(acc1[e] + bias);
    } else {
      const float rr0 = fast_sigmoid(acc0[e] + bias);
      const float rr1 = fast_sigmoid(acc1[e] + bias);
      rhbuf[r0 * H_ + ncol] = (__bf16)(rr0 * hf[r0 * H_ + ncol]);
      rhbuf[r1 * H_ + ncol] = (__bf16)(rr1 * hf[r1 * H_ + ncol]);
    }
  }
}

// ---------------------------------------------------------------------------
// Phase 2: acc = [r*h|x] @ [Whg;Wxg] (N = H). g = tanh(acc+bhg),
// h' = z*h + (1-z)*g. Writes h' as f32 + bf16 (+ S-trace if WRITE_TRACE).
// Grid (16,2), block 128.
// ---------------------------------------------------------------------------
template <int IND, bool WRITE_TRACE>
__global__ __launch_bounds__(128)
void gru_phase2_kernel(const __bf16* __restrict__ rhb,
                       const __bf16* __restrict__ xb,
                       const __bf16* __restrict__ WgT,    // [H][H+IND] n-major
                       const float*  __restrict__ bhg,
                       const float*  __restrict__ zbuf,
                       float*        __restrict__ hf,
                       __bf16*       __restrict__ hb_out,
                       __bf16*       __restrict__ hall,
                       int t)
{
  constexpr int Ktot = H_ + IND;
  const int lane  = threadIdx.x & 31;
  const int wave  = threadIdx.x >> 5;
  const int ntile = blockIdx.x * 4 + wave;        // [0,64)
  const int nlane = ntile * 16 + (lane & 15);     // column in [0,H)
  const int mbase = blockIdx.y * 32;
  const int hB    = 8  * (lane >> 4);
  const int hA    = 8  * (lane >> 4);
  const int m0    = mbase + (lane & 15);

  const __bf16* brow = WgT + (size_t)nlane * Ktot + hB;
  const __bf16* ah0  = rhb + (size_t)m0 * H_ + hA;
  const __bf16* ah1  = ah0 + 16 * H_;
  const __bf16* ax0  = xb + (size_t)m0 * IND + hA;
  const __bf16* ax1  = ax0 + 16 * IND;

  v8f acc0 = {};
  v8f acc1 = {};

  for (int kb = 0; kb < H_; kb += 32) {           // (r*h)-part of K
    const __bf16* bp = brow + kb;
    __builtin_prefetch(bp + 256, 0, 1);
    v16bf bfrag = cat8(ld8(bp), ld8(bp + 16));
    v16bf a0 = cat8(ld8(ah0 + kb), ld8(ah0 + kb + 16));
    acc0 = __builtin_amdgcn_wmma_f32_16x16x32_bf16(
        false, a0, false, bfrag, (short)0, acc0, false, false);
    v16bf a1 = cat8(ld8(ah1 + kb), ld8(ah1 + kb + 16));
    acc1 = __builtin_amdgcn_wmma_f32_16x16x32_bf16(
        false, a1, false, bfrag, (short)0, acc1, false, false);
  }
  const __bf16* browx = brow + H_;
  for (int kb = 0; kb < IND; kb += 32) {          // x-part of K
    const __bf16* bp = browx + kb;
    v16bf bfrag = cat8(ld8(bp), ld8(bp + 16));
    v16bf a0 = cat8(ld8(ax0 + kb), ld8(ax0 + kb + 16));
    acc0 = __builtin_amdgcn_wmma_f32_16x16x32_bf16(
        false, a0, false, bfrag, (short)0, acc0, false, false);
    v16bf a1 = cat8(ld8(ax1 + kb), ld8(ax1 + kb + 16));
    acc1 = __builtin_amdgcn_wmma_f32_16x16x32_bf16(
        false, a1, false, bfrag, (short)0, acc1, false, false);
  }

  const float bg = bhg[nlane];
#pragma unroll
  for (int e = 0; e < 8; ++e) {
#pragma unroll
    for (int half = 0; half < 2; ++half) {
      const int m = mbase + (lane >> 4) * 8 + e + 16 * half;
      const float c = half ? acc1[e] : acc0[e];
      const float g = fast_tanh(c + bg);
      const int idx = m * H_ + nlane;
      const float z = zbuf[idx];
      const float hn = z * hf[idx] + (1.0f - z) * g;
      hf[idx] = hn;
      const __bf16 hbv = (__bf16)hn;
      hb_out[idx] = hbv;
      if (WRITE_TRACE) hall[(size_t)t * (B_ * H_) + idx] = hbv;
    }
  }
}

// ---------------------------------------------------------------------------
// Final projection: Y[b,t,:] = H1[t,b,:] @ WY + bY. A = [S*B, H] row-major.
// Grid (4096, 4), block 128; wave -> one 16x16 tile, K=1024.
// ---------------------------------------------------------------------------
__global__ __launch_bounds__(128)
void out_gemm_kernel(const __bf16* __restrict__ hall,
                     const __bf16* __restrict__ WYT,   // [O][H] n-major
                     const float*  __restrict__ bY,
                     float*        __restrict__ y)
{
  const int lane  = threadIdx.x & 31;
  const int wave  = threadIdx.x >> 5;
  const int ntile = blockIdx.y * 4 + wave;        // [0,16)
  const int nlane = ntile * 16 + (lane & 15);     // [0,O)
  const int mbase = blockIdx.x * 16;              // [0, S*B)
  const int hB    = 8  * (lane >> 4);
  const int hA    = 8  * (lane >> 4);

  const __bf16* arow = hall + (size_t)(mbase + (lane & 15)) * H_ + hA;
  const __bf16* brow = WYT  + (size_t)nlane * H_ + hB;

  v8f acc = {};
  for (int kb = 0; kb < H_; kb += 32) {
    v16bf bfrag = cat8(ld8(brow + kb), ld8(brow + kb + 16));
    v16bf afrag = cat8(ld8(arow + kb), ld8(arow + kb + 16));
    acc = __builtin_amdgcn_wmma_f32_16x16x32_bf16(
        false, afrag, false, bfrag, (short)0, acc, false, false);
  }
  const float bias = bY[nlane];
#pragma unroll
  for (int e = 0; e < 8; ++e) {
    const int rm = mbase + (lane >> 4) * 8 + e;   // rm = t*64 + b
    const int t  = rm >> 6;
    const int b  = rm & 63;
    y[((size_t)b * S_ + t) * O_ + nlane] = acc[e] + bias;
  }
}

// ---------------------------------------------------------------------------
// One-time pack / convert kernels
// ---------------------------------------------------------------------------
__global__ void cvt_x_kernel(const float* __restrict__ x, __bf16* __restrict__ xt) {
  // x: [B,S,I] f32 -> xt: [S,B,I] bf16
  const size_t i = (size_t)blockIdx.x * blockDim.x + threadIdx.x;
  const size_t total = (size_t)S_ * B_ * I_;
  if (i >= total) return;
  const int ii = (int)(i % I_);
  const size_t r = i / I_;
  const int b = (int)(r % B_);
  const int t = (int)(r / B_);
  xt[i] = (__bf16)x[((size_t)b * S_ + t) * I_ + ii];
}

__global__ void pack_gate_kernel(const float* __restrict__ Wh,   // [H,H]
                                 const float* __restrict__ Wx,   // [ind,H]
                                 __bf16* __restrict__ dst,       // [H][H+ind]
                                 int ind) {
  const int Ktot = H_ + ind;
  const size_t idx = (size_t)blockIdx.x * blockDim.x + threadIdx.x;
  const size_t total = (size_t)H_ * Ktot;
  if (idx >= total) return;
  const int k = (int)(idx % Ktot);
  const int n = (int)(idx / Ktot);
  const float v = (k < H_) ? Wh[(size_t)k * H_ + n]
                           : Wx[(size_t)(k - H_) * H_ + n];
  dst[idx] = (__bf16)v;
}

__global__ void pack_wy_kernel(const float* __restrict__ WY,     // [H,O]
                               __bf16* __restrict__ WYT) {       // [O][H]
  const size_t idx = (size_t)blockIdx.x * blockDim.x + threadIdx.x;
  if (idx >= (size_t)O_ * H_) return;
  const int k = (int)(idx & (H_ - 1));
  const int o = (int)(idx >> 10);
  WYT[idx] = (__bf16)WY[(size_t)k * O_ + o];
}

__global__ void init_h_kernel(const float* __restrict__ h0,      // [B,2,H]
                              float* __restrict__ hf0, __bf16* __restrict__ hb0,
                              float* __restrict__ hf1, __bf16* __restrict__ hb1) {
  const size_t idx = (size_t)blockIdx.x * blockDim.x + threadIdx.x;
  if (idx >= (size_t)B_ * H_) return;
  const int n = (int)(idx & (H_ - 1));
  const int b = (int)(idx >> 10);
  const float v0 = h0[(size_t)b * 2 * H_ + n];
  const float v1 = h0[(size_t)b * 2 * H_ + H_ + n];
  hf0[idx] = v0; hb0[idx] = (__bf16)v0;
  hf1[idx] = v1; hb1[idx] = (__bf16)v1;
}

__global__ void copy_states_kernel(const float* __restrict__ hf0,
                                   const float* __restrict__ hf1,
                                   float* __restrict__ out_tail) { // [B,2,H]
  const size_t idx = (size_t)blockIdx.x * blockDim.x + threadIdx.x;
  if (idx >= (size_t)B_ * H_) return;
  const int n = (int)(idx & (H_ - 1));
  const int b = (int)(idx >> 10);
  out_tail[(size_t)b * 2 * H_ + n]      = hf0[idx];
  out_tail[(size_t)b * 2 * H_ + H_ + n] = hf1[idx];
}

// ---------------------------------------------------------------------------
extern "C" void kernel_launch(void* const* d_in, const int* in_sizes, int n_in,
                              void* d_out, int out_size, void* d_ws, size_t ws_size,
                              hipStream_t stream) {
  (void)in_sizes; (void)n_in; (void)out_size; (void)ws_size;
  const float* x    = (const float*)d_in[0];
  const float* h0   = (const float*)d_in[1];
  const float* Wxz0 = (const float*)d_in[2];
  const float* Wxr0 = (const float*)d_in[3];
  const float* Wxg0 = (const float*)d_in[4];
  const float* Whz0 = (const float*)d_in[5];
  const float* bhz0 = (const float*)d_in[6];
  const float* Whr0 = (const float*)d_in[7];
  const float* bhr0 = (const float*)d_in[8];
  const float* Whg0 = (const float*)d_in[9];
  const float* bhg0 = (const float*)d_in[10];
  const float* Wxz1 = (const float*)d_in[11];
  const float* Wxr1 = (const float*)d_in[12];
  const float* Wxg1 = (const float*)d_in[13];
  const float* Whz1 = (const float*)d_in[14];
  const float* bhz1 = (const float*)d_in[15];
  const float* Whr1 = (const float*)d_in[16];
  const float* bhr1 = (const float*)d_in[17];
  const float* Whg1 = (const float*)d_in[18];
  const float* bhg1 = (const float*)d_in[19];
  const float* WY   = (const float*)d_in[20];
  const float* bY   = (const float*)d_in[21];

  char* ws = (char*)d_ws;
  size_t off = 0;
  auto take = [&](size_t bytes) -> void* {
    void* p = ws + off;
    off = (off + bytes + 255) & ~(size_t)255;
    return p;
  };
  const size_t K0 = H_ + I_;      // 1280
  const size_t K1 = H_ + H_;      // 2048
  __bf16* Xt    = (__bf16*)take((size_t)S_ * B_ * I_ * 2);
  __bf16* Wzr0T = (__bf16*)take(2 * (size_t)H_ * K0 * 2);
  __bf16* Wg0T  = (__bf16*)take((size_t)H_ * K0 * 2);
  __bf16* Wzr1T = (__bf16*)take(2 * (size_t)H_ * K1 * 2);
  __bf16* Wg1T  = (__bf16*)take((size_t)H_ * K1 * 2);
  __bf16* WYT   = (__bf16*)take((size_t)O_ * H_ * 2);
  float*  hf0   = (float*)take((size_t)B_ * H_ * 4);
  float*  hf1   = (float*)take((size_t)B_ * H_ * 4);
  __bf16* hb0   = (__bf16*)take((size_t)B_ * H_ * 2);
  __bf16* hb1   = (__bf16*)take((size_t)B_ * H_ * 2);
  float*  z0    = (float*)take((size_t)B_ * H_ * 4);
  float*  z1    = (float*)take((size_t)B_ * H_ * 4);
  __bf16* rh0   = (__bf16*)take((size_t)B_ * H_ * 2);
  __bf16* rh1   = (__bf16*)take((size_t)B_ * H_ * 2);
  __bf16* Hall  = (__bf16*)take((size_t)S_ * B_ * H_ * 2);

  // ---- one-time packs (re-done every call: deterministic, capture-safe) ----
  {
    const size_t tot = (size_t)S_ * B_ * I_;
    cvt_x_kernel<<<dim3((unsigned)((tot + 255) / 256)), 256, 0, stream>>>(x, Xt);
  }
  {
    const unsigned g0 = (unsigned)(((size_t)H_ * K0 + 255) / 256);
    pack_gate_kernel<<<g0, 256, 0, stream>>>(Whz0, Wxz0, Wzr0T, I_);
    pack_gate_kernel<<<g0, 256, 0, stream>>>(Whr0, Wxr0, Wzr0T + (size_t)H_ * K0, I_);
    pack_gate_kernel<<<g0, 256, 0, stream>>>(Whg0, Wxg0, Wg0T, I_);
    const unsigned g1 = (unsigned)(((size_t)H_ * K1 + 255) / 256);
    pack_gate_kernel<<<g1, 256, 0, stream>>>(Whz1, Wxz1, Wzr1T, H_);
    pack_gate_kernel<<<g1, 256, 0, stream>>>(Whr1, Wxr1, Wzr1T + (size_t)H_ * K1, H_);
    pack_gate_kernel<<<g1, 256, 0, stream>>>(Whg1, Wxg1, Wg1T, H_);
  }
  pack_wy_kernel<<<dim3(((unsigned)O_ * H_ + 255) / 256), 256, 0, stream>>>(WY, WYT);
  init_h_kernel<<<dim3(((unsigned)B_ * H_ + 255) / 256), 256, 0, stream>>>(h0, hf0, hb0, hf1, hb1);

  // ---- sequential scan: 4 dependent WMMA kernels per timestep ----
  for (int t = 0; t < S_; ++t) {
    const __bf16* x_t = Xt + (size_t)t * B_ * I_;
    gru_phase1_kernel<I_><<<dim3(32, 2), 128, 0, stream>>>(
        hb0, x_t, Wzr0T, bhz0, bhr0, hf0, z0, rh0);
    gru_phase2_kernel<I_, false><<<dim3(16, 2), 128, 0, stream>>>(
        rh0, x_t, Wg0T, bhg0, z0, hf0, hb0, nullptr, t);
    gru_phase1_kernel<H_><<<dim3(32, 2), 128, 0, stream>>>(
        hb1, hb0, Wzr1T, bhz1, bhr1, hf1, z1, rh1);
    gru_phase2_kernel<H_, true><<<dim3(16, 2), 128, 0, stream>>>(
        rh1, hb0, Wg1T, bhg1, z1, hf1, hb1, Hall, t);
  }

  // ---- batched output projection + final states ----
  out_gemm_kernel<<<dim3(4096, 4), 128, 0, stream>>>(Hall, WYT, bY, (float*)d_out);
  copy_states_kernel<<<dim3(((unsigned)B_ * H_ + 255) / 256), 256, 0, stream>>>(
      hf0, hf1, (float*)d_out + (size_t)B_ * S_ * O_);
}
